// ScaledDotProductAttentionCustom_33887291965754
// MI455X (gfx1250) — matrix operand order repaired and checked by reading, not compiled, not run on hardware
//
#include <hip/hip_runtime.h>
#include <hip/hip_bf16.h>

// ---------------------------------------------------------------------------
// ScaledDotProductAttentionCustom for MI455X (gfx1250, wave32, WMMA bf16)
//   B=16, S=2048, D=64.  Output = [out (B*S*D) | attention (B*S*S)] fp32.
// Store-bound: attention write = 268MB (~11.5us @ 23.3TB/s floor).
// Single-pass: score tile (16 rows x 2048 cols) lives in LDS, softmax in
// place, attention streamed out once with NT stores (no L2 pollution so the
// per-batch K/V^T working sets stay L2-resident), out-tile via 2nd WMMA pass.
// ---------------------------------------------------------------------------

typedef __attribute__((ext_vector_type(16))) __bf16 v16bf;
typedef __attribute__((ext_vector_type(8)))  float  v8f;
typedef __attribute__((ext_vector_type(4)))  float  v4f;   // native vec for NT stores

#define B_ 16
#define S_ 2048
#define D_ 64
#define ROWS 16
#define SCPAD 2064          // 2048 + 16 pad floats -> no LDS bank conflicts
#define NTHREADS 256        // 8 waves (wave32)

static __device__ __forceinline__ unsigned short f32_to_bf16(float f) {
    unsigned int u = __float_as_uint(f);
    unsigned int lsb = (u >> 16) & 1u;
    u += 0x7fffu + lsb;                      // round-to-nearest-even
    return (unsigned short)(u >> 16);
}

// Build a v16bf operand from two 16-byte (8 x bf16) chunks.
static __device__ __forceinline__ v16bf load_v16bf_2(const unsigned short* p0,
                                                     const unsigned short* p1) {
    union { uint4 q[2]; v16bf v; } u;
    u.q[0] = *reinterpret_cast<const uint4*>(p0);
    u.q[1] = *reinterpret_cast<const uint4*>(p1);
    return u.v;
}

// ---------------------------------------------------------------------------
// Kernel 0: fp32 -> bf16 conversion; V additionally transposed to [B, D, S]
// so the WMMA B-operand (K-dim = token j) is contiguous per lane.
// ---------------------------------------------------------------------------
__global__ __launch_bounds__(256) void convert_kernel(
    const float* __restrict__ Q, const float* __restrict__ K,
    const float* __restrict__ V,
    unsigned short* __restrict__ Qbf, unsigned short* __restrict__ Kbf,
    unsigned short* __restrict__ VbfT) {
    size_t idx = (size_t)blockIdx.x * blockDim.x + threadIdx.x;
    if (idx >= (size_t)B_ * S_ * D_) return;
    Qbf[idx] = f32_to_bf16(Q[idx]);
    Kbf[idx] = f32_to_bf16(K[idx]);
    int d = (int)(idx % D_);
    int s = (int)((idx / D_) % S_);
    int b = (int)(idx / ((size_t)S_ * D_));
    VbfT[((size_t)b * D_ + d) * S_ + s] = f32_to_bf16(V[idx]);
}

// ---------------------------------------------------------------------------
// Kernel 1: diag_s = |q_s . k_s| / 8 ; idw = softmax(diag) per batch.
// One block per batch.
// ---------------------------------------------------------------------------
__global__ __launch_bounds__(256) void idw_kernel(
    const float* __restrict__ Q, const float* __restrict__ K,
    float* __restrict__ idw) {
    __shared__ float diag[S_];
    __shared__ float red[256];
    const int b = blockIdx.x;
    const int tid = threadIdx.x;

    float lmax = -1e30f;
    for (int s = tid; s < S_; s += 256) {
        const float4* q4 = reinterpret_cast<const float4*>(Q + ((size_t)(b * S_ + s)) * D_);
        const float4* k4 = reinterpret_cast<const float4*>(K + ((size_t)(b * S_ + s)) * D_);
        float dot = 0.f;
        #pragma unroll
        for (int i = 0; i < D_ / 4; ++i) {
            float4 a = q4[i], c = k4[i];
            dot += a.x * c.x + a.y * c.y + a.z * c.z + a.w * c.w;
        }
        float dv = fabsf(dot) * 0.125f;
        diag[s] = dv;
        lmax = fmaxf(lmax, dv);
    }
    red[tid] = lmax;
    __syncthreads();
    for (int o = 128; o > 0; o >>= 1) {
        if (tid < o) red[tid] = fmaxf(red[tid], red[tid + o]);
        __syncthreads();
    }
    float mx = red[0];
    __syncthreads();

    float lsum = 0.f;
    for (int s = tid; s < S_; s += 256) lsum += __expf(diag[s] - mx);
    red[tid] = lsum;
    __syncthreads();
    for (int o = 128; o > 0; o >>= 1) {
        if (tid < o) red[tid] += red[tid + o];
        __syncthreads();
    }
    float inv = 1.f / red[0];
    for (int s = tid; s < S_; s += 256)
        idw[(size_t)b * S_ + s] = __expf(diag[s] - mx) * inv;
}

// ---------------------------------------------------------------------------
// Kernel 2: main attention. One workgroup = 16 query rows of one batch.
// ---------------------------------------------------------------------------
__global__ __launch_bounds__(NTHREADS) void attn_kernel(
    const float* __restrict__ idw,            // [B,S]
    const unsigned short* __restrict__ Qbf,   // [B,S,D] bf16
    const unsigned short* __restrict__ Kbf,   // [B,S,D] bf16
    const unsigned short* __restrict__ VbfT,  // [B,D,S] bf16
    float* __restrict__ outp,                 // [B,S,D]
    float* __restrict__ attnp) {              // [B,S,S]
    __shared__ float sc[ROWS * SCPAD];        // 129 KB  score / p tile
    __shared__ float idwcol[S_];              // 8 KB    idw_j
    __shared__ unsigned short qtile[ROWS * D_]; // 2 KB  Q tile (bf16)
    __shared__ float rowscale[ROWS];
    __shared__ float outred[ROWS * D_];       // 4 KB    out reduction

    const int tid  = threadIdx.x;
    const int lane = tid & 31;
    const int wave = tid >> 5;                // 0..7
    const int g    = lane >> 4;               // half-wave group 0/1
    const int ln   = lane & 15;
    const int b       = blockIdx.y;
    const int rowbase = blockIdx.x * ROWS;

    // -------- Phase 0: stage Q tile + idw column vector, zero reduction buf
    {
        const uint2* qsrc =
            reinterpret_cast<const uint2*>(Qbf + ((size_t)(b * S_ + rowbase)) * D_);
        reinterpret_cast<uint2*>(qtile)[tid] = qsrc[tid];      // 2 KB

        const float4* isrc = reinterpret_cast<const float4*>(idw + (size_t)b * S_);
        float4* idst = reinterpret_cast<float4*>(idwcol);
        for (int i = tid; i < S_ / 4; i += NTHREADS) idst[i] = isrc[i];

        reinterpret_cast<float4*>(outred)[tid] = make_float4(0.f, 0.f, 0.f, 0.f);
    }
    __syncthreads();

    // -------- Phase 1: scores = |Q K^T| / 8 via v_wmma_f32_16x16x32_bf16
    {
        // A operand (Q tile): lane holds M=ln; K chunks at kb+8g and kb+16+8g
        const unsigned short* qrow = qtile + ln * D_;
        for (int ct = 0; ct < 16; ++ct) {                 // 16 col tiles / wave
            const int colbase = (wave * 16 + ct) * 16;
            const int t = colbase + ln;                   // key token (B: N=ln)
            const unsigned short* krow = Kbf + ((size_t)(b * S_ + t)) * D_;
            v8f acc = {};
            #pragma unroll
            for (int kk = 0; kk < 2; ++kk) {              // D=64 -> 2 k-steps
                const int kb = kk * 32;
                v16bf a  = load_v16bf_2(qrow + kb + 8 * g, qrow + kb + 16 + 8 * g);
                v16bf bm = load_v16bf_2(krow + kb + 8 * g, krow + kb + 16 + 8 * g);
                acc = __builtin_amdgcn_wmma_f32_16x16x32_bf16(
                    false, a, false, bm, (short)0, acc, false, false);
            }
            #pragma unroll
            for (int r = 0; r < 8; ++r) {                 // D layout: M=r+8g, N=ln
                const int m = r + 8 * g;
                sc[m * SCPAD + colbase + ln] = fabsf(acc[r]) * 0.125f;
            }
        }
    }
    __syncthreads();

    // -------- Phase 2: row softmax in place; rowscale = idw_i / sum_i
    {
        for (int rr = 0; rr < 2; ++rr) {                  // 8 waves x 2 rows
            const int m = wave * 2 + rr;
            float* row = sc + m * SCPAD;
            float mx = -1e30f;
            for (int c = lane; c < S_; c += 32) mx = fmaxf(mx, row[c]);
            #pragma unroll
            for (int o = 16; o > 0; o >>= 1) mx = fmaxf(mx, __shfl_xor(mx, o, 32));
            float sum = 0.f;
            for (int c = lane; c < S_; c += 32) {
                float e = __expf(row[c] - mx);
                row[c] = e;                               // keep p_ij
                sum += e;
            }
            #pragma unroll
            for (int o = 16; o > 0; o >>= 1) sum += __shfl_xor(sum, o, 32);
            if (lane == 0) rowscale[m] = idwcol[rowbase + m] / sum;
        }
    }
    __syncthreads();

    // -------- Phase 3a: stream attention tile (the 268MB write).
    // Non-temporal stores: this data is write-once, never re-read; keep the
    // per-batch K/V^T working sets resident in the 192MB L2 instead.
    {
        const size_t abase = ((size_t)b * S_ + rowbase) * S_;
        v4f* adst = reinterpret_cast<v4f*>(attnp + abase);
        for (int i = tid; i < ROWS * (S_ / 4); i += NTHREADS) {
            const int m  = i >> 9;                        // / (S_/4)
            const int c4 = i & 511;
            v4f p = reinterpret_cast<const v4f*>(sc + m * SCPAD)[c4];
            v4f w = reinterpret_cast<const v4f*>(idwcol)[c4];
            v4f o = rowscale[m] * p * w;
            __builtin_nontemporal_store(o, &adst[m * (S_ / 4) + c4]);
        }
    }

    // -------- Phase 3b: out = (p * idw_j) @ V via WMMA, rowscale at the end
    {
        v8f acc[4] = {};                                  // 4 n-tiles of D=64
        const int jw = wave * 256;                        // this wave's j range
        for (int ks = 0; ks < 8; ++ks) {
            const int kb = jw + ks * 32;
            union { unsigned short h[16]; v16bf v; } ua;  // A: attention bf16
            {
                const float* prow = sc + ln * SCPAD;      // M = ln
                const int c0 = kb + 8 * g;
                const int c1 = kb + 16 + 8 * g;
                #pragma unroll
                for (int e = 0; e < 8; ++e)
                    ua.h[e] = f32_to_bf16(prow[c0 + e] * idwcol[c0 + e]);
                #pragma unroll
                for (int e = 0; e < 8; ++e)
                    ua.h[8 + e] = f32_to_bf16(prow[c1 + e] * idwcol[c1 + e]);
            }
            const int j0 = kb + 8 * g;                    // B: K chunks at j0 and j0+16
            #pragma unroll
            for (int nt = 0; nt < 4; ++nt) {
                const int d = nt * 16 + ln;               // B: N = ln
                const unsigned short* vp =
                    VbfT + ((size_t)(b * D_ + d)) * S_ + j0;
                v16bf bm = load_v16bf_2(vp, vp + 16);
                acc[nt] = __builtin_amdgcn_wmma_f32_16x16x32_bf16(
                    false, ua.v, false, bm, (short)0, acc[nt], false, false);
            }
        }
        // cross-wave reduction of the 16x64 partials via ds_add_f32
        #pragma unroll
        for (int nt = 0; nt < 4; ++nt) {
            #pragma unroll
            for (int r = 0; r < 8; ++r) {
                const int m = r + 8 * g;
                const int d = nt * 16 + ln;
                atomicAdd(&outred[m * D_ + d], acc[nt][r]);
            }
        }
    }
    __syncthreads();

    // -------- Final: out[b, rowbase+m, :] = rowscale[m] * outred[m, :]
    {
        const int m = tid >> 4;                           // 16 float4 per row
        v4f v = reinterpret_cast<const v4f*>(outred)[tid] * rowscale[m];
        v4f* odst =
            reinterpret_cast<v4f*>(outp + ((size_t)(b * S_ + rowbase)) * D_);
        __builtin_nontemporal_store(v, &odst[tid]);
    }
}

// ---------------------------------------------------------------------------
extern "C" void kernel_launch(void* const* d_in, const int* in_sizes, int n_in,
                              void* d_out, int out_size, void* d_ws, size_t ws_size,
                              hipStream_t stream) {
    const float* Q = (const float*)d_in[0];
    const float* K = (const float*)d_in[1];
    const float* V = (const float*)d_in[2];

    float* outp  = (float*)d_out;                       // [B,S,D]
    float* attnp = outp + (size_t)B_ * S_ * D_;         // [B,S,S]

    // workspace: idw (128KB) | Qbf (4MB) | Kbf (4MB) | VbfT (4MB)
    char* ws = (char*)d_ws;
    float* idw = (float*)ws;
    unsigned short* Qbf  = (unsigned short*)(ws + (size_t)B_ * S_ * sizeof(float));
    unsigned short* Kbf  = Qbf + (size_t)B_ * S_ * D_;
    unsigned short* VbfT = Kbf + (size_t)B_ * S_ * D_;

    const int total = B_ * S_ * D_;
    convert_kernel<<<(total + 255) / 256, 256, 0, stream>>>(Q, K, V, Qbf, Kbf, VbfT);
    idw_kernel<<<B_, 256, 0, stream>>>(Q, K, idw);

    dim3 grid(S_ / ROWS, B_);
    attn_kernel<<<grid, NTHREADS, 0, stream>>>(idw, Qbf, Kbf, VbfT, outp, attnp);
}
